// SteerableNCA_75823352643879
// MI455X (gfx1250) — matrix operand order repaired and checked by reading, not compiled, not run on hardware
//
#include <hip/hip_runtime.h>
#include <hip/hip_bf16.h>

typedef __attribute__((ext_vector_type(16))) _Float16 v16h;
typedef __attribute__((ext_vector_type(8)))  _Float16 v8h;
typedef __attribute__((ext_vector_type(8)))  float    v8f;

#define BB      4
#define HH      128
#define WW      128
#define CC      17          // channels
#define PCP     64          // padded 3*C = 51 -> 64
#define WID     128         // hidden width
#define NSTEPS  48
#define TX      16          // pixels per wave (N dim of WMMA)
#define TY      4           // rows per block (one wave per row)
#define SLICE   (BB*CC*HH*WW)       // one dev_path step slice (floats)
#define FINAL_SZ (BB*4*HH*WW)

// ---------------------------------------------------------------------------
// Kernel A: per-step heavy work. Builds perception vector p (state, px, py),
// runs both dense layers via v_wmma_f32_16x16x32_f16, writes
// pre = state + dx * (rnd < 0.5) to workspace.  One wave handles 16 pixels
// of one row; block = 4 waves = 16x4 pixel tile.
// ---------------------------------------------------------------------------
__global__ __launch_bounds__(128)
void nca_step_a(const float* __restrict__ state,  // [B,C,H,W]
                const float* __restrict__ W1,     // [128,51]
                const float* __restrict__ b1,     // [128]
                const float* __restrict__ W2,     // [17,128]
                const float* __restrict__ b2,     // [17]
                const float* __restrict__ rnd,    // [B,H,W] (this step)
                float* __restrict__ pre)          // [B,C,H,W]
{
    __shared__ __align__(16) float    s_tile[CC][TY+2][TX+2]; // halo state tile
    __shared__ __align__(16) _Float16 w1s[WID][PCP];          // W1 f16, k-padded
    __shared__ __align__(16) _Float16 w2s[32][WID];           // W2 f16, M-padded
    __shared__ __align__(16) float    b1s[WID];
    __shared__ __align__(16) float    b2s[32];
    __shared__ __align__(16) _Float16 pl[TY][TX][PCP];        // p, per wave
    __shared__ __align__(16) _Float16 hl[TY][TX][WID];        // h, per wave

    const int tid  = threadIdx.x;
    const int wv   = tid >> 5;        // wave = row in tile
    const int ln   = tid & 31;
    const int half = ln >> 4;         // 0: lanes 0-15, 1: lanes 16-31
    const int nb   = ln & 15;         // pixel (N) / matrix row (M) index
    const int b    = blockIdx.z;
    const int x0   = blockIdx.x * TX;
    const int y0   = blockIdx.y * TY;

    // ---- stage weights into LDS (f16) ----
    for (int i = tid; i < WID*PCP; i += 128) {
        int o = i >> 6, k = i & 63;
        w1s[o][k] = (k < 51) ? (_Float16)W1[o*51 + k] : (_Float16)0.f;
    }
    for (int i = tid; i < 32*WID; i += 128) {
        int o = i >> 7, k = i & 127;
        w2s[o][k] = (o < CC) ? (_Float16)W2[o*WID + k] : (_Float16)0.f;
    }
    if (tid < WID) b1s[tid] = b1[tid];
    if (tid < 32)  b2s[tid] = (tid < CC) ? b2[tid] : 0.f;

    // ---- stage state halo tile (zero padded, matches conv 'SAME') ----
    for (int i = tid; i < CC*(TY+2)*(TX+2); i += 128) {
        int c  = i / ((TY+2)*(TX+2));
        int r  = (i / (TX+2)) % (TY+2);
        int xx = i % (TX+2);
        int gy = y0 + r - 1, gx = x0 + xx - 1;
        float v = 0.f;
        if (gy >= 0 && gy < HH && gx >= 0 && gx < WW)
            v = state[((b*CC + c)*HH + gy)*WW + gx];
        s_tile[c][r][xx] = v;
    }
    __syncthreads();

    // ---- build p = [state, px, py] (f16) for this wave's 16 pixels ----
    const int row = wv;
    {
        float ang = s_tile[CC-1][row+1][nb+1];
        float ca = cosf(ang), sa = sinf(ang);
        for (int c = half; c < CC; c += 2) {     // halves split channels
            float s00 = s_tile[c][row  ][nb  ], s01 = s_tile[c][row  ][nb+1], s02 = s_tile[c][row  ][nb+2];
            float s10 = s_tile[c][row+1][nb  ],                               s12 = s_tile[c][row+1][nb+2];
            float s20 = s_tile[c][row+2][nb  ], s21 = s_tile[c][row+2][nb+1], s22 = s_tile[c][row+2][nb+2];
            // cross-correlation with SOBEL_X / SOBEL_Y (no kernel flip), /8
            float gx = (s02 - s00 + 2.f*(s12 - s10) + s22 - s20) * 0.125f;
            float gy = (s20 - s00 + 2.f*(s21 - s01) + s22 - s02) * 0.125f;
            pl[wv][nb][c]      = (_Float16)s_tile[c][row+1][nb+1];
            pl[wv][nb][17 + c] = (_Float16)( ca*gx + sa*gy);   // px
            pl[wv][nb][34 + c] = (_Float16)( ca*gy - sa*gx);   // py
        }
        if (half == 0) {
            for (int k = 51; k < PCP; ++k) pl[wv][nb][k] = (_Float16)0.f;
        }
    }
    __syncthreads();

    // A-fragment lane K split: lanes 0-15 -> 0-15, lanes 16-31 -> 16-31
    // B-fragment lane K split: lanes 0-15 -> 0-15, lanes 16-31 -> 16-31
    const int kA = 16 * half;
    const int kB = 16 * half;

    // ---- layer 1: h = relu(W1 . p + b1); K = 64 (2 chunks), 8 M-tiles ----
    #pragma unroll
    for (int t = 0; t < 8; ++t) {
        const _Float16* w1row = &w1s[16*t + nb][0];
        v8f acc = {};
        #pragma unroll
        for (int ch = 0; ch < 2; ++ch) {
            union { v16h v; v8h h[2]; } af, bf;
            af.h[0] = *(const v8h*)(w1row + 32*ch + kA);
        af.h[1] = *(const v8h*)(w1row + 32*ch + kA + 8);
            const _Float16* prow = &pl[wv][nb][32*ch + kB];
            bf.h[0] = *(const v8h*)(prow);
            bf.h[1] = *(const v8h*)(prow + 8);
            acc = __builtin_amdgcn_wmma_f32_16x16x32_f16(
                      false, af.v, false, bf.v, (short)0, acc, false, false);
        }
        v8h hv;
        #pragma unroll
        for (int v = 0; v < 8; ++v) {
            int m = 16*t + 8*half + v;           // C/D layout: M = v + 8*half
            float h = acc[v] + b1s[m];
            hv[v] = (_Float16)(h > 0.f ? h : 0.f);
        }
        *(v8h*)&hl[wv][nb][16*t + 8*half] = hv;
    }
    __syncthreads();

    // ---- layer 2: dx = W2 . h + b2; K = 128 (4 chunks), 2 M-tiles ----
    const int gx_ = x0 + nb, gy_ = y0 + row;
    const float r   = rnd[(b*HH + gy_)*WW + gx_];
    const float msk = (r < 0.5f) ? 1.f : 0.f;
    #pragma unroll
    for (int t2 = 0; t2 < 2; ++t2) {
        const _Float16* w2row = &w2s[16*t2 + nb][0];
        v8f acc = {};
        #pragma unroll
        for (int ch = 0; ch < 4; ++ch) {
            union { v16h v; v8h h[2]; } af, bf;
            af.h[0] = *(const v8h*)(w2row + 32*ch + kA);
            af.h[1] = *(const v8h*)(w2row + 32*ch + kA + 8);
            const _Float16* hrow = &hl[wv][nb][32*ch + kB];
            bf.h[0] = *(const v8h*)(hrow);
            bf.h[1] = *(const v8h*)(hrow + 8);
            acc = __builtin_amdgcn_wmma_f32_16x16x32_f16(
                      false, af.v, false, bf.v, (short)0, acc, false, false);
        }
        #pragma unroll
        for (int v = 0; v < 8; ++v) {
            int m = 16*t2 + 8*half + v;
            if (m < CC) {
                float dx = acc[v] + b2s[m];
                float nw = s_tile[m][row+1][nb+1] + dx * msk;
                pre[((b*CC + m)*HH + gy_)*WW + gx_] = nw;
            }
        }
    }
}

// ---------------------------------------------------------------------------
// Kernel B: alive gating.  pre/post 3x3 maxpool ('SAME', -inf padding) on
// channel 3; new = pre * alive; writes this step's dev_path slice (which is
// also the next step's state).
// ---------------------------------------------------------------------------
__global__ __launch_bounds__(256)
void nca_step_b(const float* __restrict__ state,  // old state [B,C,H,W]
                const float* __restrict__ pre,    // [B,C,H,W]
                float* __restrict__ out)          // dev_path slice [B,C,H,W]
{
    int idx = blockIdx.x * 256 + threadIdx.x;
    if (idx >= BB*HH*WW) return;
    int x = idx % WW, y = (idx / WW) % HH, b = idx / (HH*WW);
    const float* sA = &state[(size_t)(b*CC + 3)*HH*WW];
    const float* pA = &pre  [(size_t)(b*CC + 3)*HH*WW];
    float mpre = -3.4e38f, mpost = -3.4e38f;
    #pragma unroll
    for (int dy = -1; dy <= 1; ++dy) {
        int yy = y + dy; if (yy < 0 || yy >= HH) continue;
        #pragma unroll
        for (int dx = -1; dx <= 1; ++dx) {
            int xx = x + dx; if (xx < 0 || xx >= WW) continue;
            mpre  = fmaxf(mpre,  sA[yy*WW + xx]);
            mpost = fmaxf(mpost, pA[yy*WW + xx]);
        }
    }
    float alive = (mpre > 0.1f && mpost > 0.1f) ? 1.f : 0.f;
    for (int c = 0; c < CC; ++c) {
        size_t o = ((size_t)(b*CC + c)*HH + y)*WW + x;
        out[o] = pre[o] * alive;
    }
}

// ---------------------------------------------------------------------------
// Kernel C: final[:, :4] = last step's first 4 channels.
// ---------------------------------------------------------------------------
__global__ __launch_bounds__(256)
void nca_final(const float* __restrict__ last, float* __restrict__ fin)
{
    int i = blockIdx.x * 256 + threadIdx.x;
    if (i >= FINAL_SZ) return;
    int x = i % WW, y = (i / WW) % HH, c = (i / (HH*WW)) % 4, b = i / (4*HH*WW);
    fin[i] = last[((size_t)(b*CC + c)*HH + y)*WW + x];
}

// ---------------------------------------------------------------------------
extern "C" void kernel_launch(void* const* d_in, const int* in_sizes, int n_in,
                              void* d_out, int out_size, void* d_ws, size_t ws_size,
                              hipStream_t stream) {
    const float* init = (const float*)d_in[0];   // [B,C,H,W]
    const float* rand = (const float*)d_in[1];   // [STEPS,B,1,H,W]
    const float* W1   = (const float*)d_in[2];   // [128,51]
    const float* b1   = (const float*)d_in[3];   // [128]
    const float* W2   = (const float*)d_in[4];   // [17,128]
    const float* b2   = (const float*)d_in[5];   // [17]

    float* fin  = (float*)d_out;                 // [B,4,H,W]
    float* path = fin + FINAL_SZ;                // [STEPS,B,C,H,W]
    float* pre  = (float*)d_ws;                  // [B,C,H,W] scratch

    const dim3 gridA(WW/TX, HH/TY, BB);          // (8, 32, 4)
    const int  nPix = BB*HH*WW;

    for (int t = 0; t < NSTEPS; ++t) {
        const float* st = (t == 0) ? init : (path + (size_t)(t-1)*SLICE);
        nca_step_a<<<gridA, 128, 0, stream>>>(st, W1, b1, W2, b2,
                                              rand + (size_t)t*BB*HH*WW, pre);
        nca_step_b<<<(nPix + 255)/256, 256, 0, stream>>>(st, pre,
                                              path + (size_t)t*SLICE);
    }
    nca_final<<<(FINAL_SZ + 255)/256, 256, 0, stream>>>(
        path + (size_t)(NSTEPS-1)*SLICE, fin);
}
